// attentionDecoderRNN_22316650070582
// MI455X (gfx1250) — matrix operand order repaired and checked by reading, not compiled, not run on hardware
//
#include <hip/hip_runtime.h>

typedef __attribute__((ext_vector_type(16))) _Float16 v16h;
typedef __attribute__((ext_vector_type(8)))  float    v8f;

constexpr int T = 64, B = 128, S = 128, H = 200, E = 200, A = 100, V = 10003;
constexpr int KP = 224;      // K=H padded to 7*32 for wmma f16 k-steps
constexpr int NPAD = 10048;  // V padded to multiple of 64 (block N-tile)

// Fragment-order offset for a 16-bit A operand element k within a row:
// per ISA 7.12.2, lane-half hl owns k in {hl*8..hl*8+7} U {16+hl*8..16+hl*8+7}.
// We store row-major rows pre-permuted so each (k-tile, hl) is 16 contiguous halves.
__device__ __forceinline__ int afrag_off(int k) {
  const int kt = k >> 5, r = k & 31;
  const int hl = (r >> 3) & 1;
  const int e  = (r & 7) + ((r >> 4) << 3);
  return kt * 32 + hl * 16 + e;
}

// ---------------------------------------------------------------------------
// K0: embedding gather + ah3 = embedded @ W3.T + b3   (one block per (t,b))
// ---------------------------------------------------------------------------
__global__ void embed_ah3_kernel(const int* __restrict__ tokens,
                                 const float* __restrict__ emb,
                                 const float* __restrict__ W3,
                                 const float* __restrict__ b3,
                                 float* __restrict__ embedded,
                                 float* __restrict__ ah3) {
  __shared__ float er[E];
  const int tb = blockIdx.x, tid = threadIdx.x;  // 128 threads
  const int tok = tokens[tb];
  for (int e = tid; e < E; e += 128) {
    float v = emb[(size_t)tok * E + e];
    er[e] = v;
    embedded[(size_t)tb * E + e] = v;
  }
  __syncthreads();
  if (tid < A) {
    const float* w = W3 + (size_t)tid * E;
    float acc = b3[tid];
    for (int e = 0; e < E; ++e) acc += w[e] * er[e];
    ah3[(size_t)tb * A + tid] = acc;
  }
}

// ---------------------------------------------------------------------------
// K1: ah2[b,s,:] = enc[b,s,:] @ W2.T + b2    (one block per (b,s))
// encoder_outputs is [S,B,H]; enc[b,s,h] = encoder_outputs[s,b,h]
// ---------------------------------------------------------------------------
__global__ void enc_ah2_kernel(const float* __restrict__ enc,
                               const float* __restrict__ W2,
                               const float* __restrict__ b2,
                               float* __restrict__ ah2) {
  __shared__ float er[H];
  const int b = blockIdx.x, s = blockIdx.y, tid = threadIdx.x;  // 128 threads
  for (int h = tid; h < H; h += 128) er[h] = enc[((size_t)s * B + b) * H + h];
  __syncthreads();
  if (tid < A) {
    const float* w = W2 + (size_t)tid * H;
    float acc = b2[tid];
    for (int h = 0; h < H; ++h) acc += w[h] * er[h];
    ah2[((size_t)b * S + s) * A + tid] = acc;
  }
}

// ---------------------------------------------------------------------------
// K2 (per step t): attention -> softmax -> context; builds x = [emb_t, ctx]
// one block per batch element, 128 threads (thread == source position s)
// ---------------------------------------------------------------------------
__global__ void attn_step_kernel(const float* __restrict__ hprev,   // [B,H]
                                 const float* __restrict__ emb_t,   // [B,E]
                                 const float* __restrict__ ah3_t,   // [B,A]
                                 const float* __restrict__ ah2,     // [B,S,A]
                                 const float* __restrict__ enc,     // [S,B,H]
                                 const float* __restrict__ W1,
                                 const float* __restrict__ b1,
                                 const float* __restrict__ attn_W,  // [A]
                                 const float* __restrict__ attn_b,  // [1]
                                 float* __restrict__ xbuf,          // [B,E+H]
                                 float* __restrict__ attn_out) {    // [B,S]
  __shared__ float hrow[H], ah1s[A], ah3s[A], lg[S], red[128];
  const int b = blockIdx.x, tid = threadIdx.x;  // 128 threads
  for (int h = tid; h < H; h += 128) hrow[h] = hprev[(size_t)b * H + h];
  if (tid < A) ah3s[tid] = ah3_t[(size_t)b * A + tid];
  __syncthreads();
  if (tid < A) {
    const float* w = W1 + (size_t)tid * H;
    float acc = b1[tid];
    for (int h = 0; h < H; ++h) acc += hrow[h] * w[h];
    ah1s[tid] = acc;
  }
  __syncthreads();
  {  // logits over A with relu
    const float* a2 = ah2 + ((size_t)b * S + tid) * A;
    float acc = attn_b[0];
    for (int a = 0; a < A; ++a) {
      float v = ah1s[a] + a2[a] + ah3s[a];
      v = v > 0.f ? v : 0.f;
      acc += attn_W[a] * v;
    }
    lg[tid] = acc;
  }
  __syncthreads();
  red[tid] = lg[tid];
  __syncthreads();
  for (int o = 64; o > 0; o >>= 1) {
    if (tid < o) red[tid] = fmaxf(red[tid], red[tid + o]);
    __syncthreads();
  }
  const float mx = red[0];
  __syncthreads();
  const float ex = __expf(lg[tid] - mx);
  red[tid] = ex;
  __syncthreads();
  for (int o = 64; o > 0; o >>= 1) {
    if (tid < o) red[tid] += red[tid + o];
    __syncthreads();
  }
  const float a_s = ex / red[0];
  lg[tid] = a_s;                              // reuse lg as softmax weights
  attn_out[(size_t)b * S + tid] = a_s;
  __syncthreads();
  for (int h = tid; h < H; h += 128) {        // ctx = a @ enc[b]
    float acc = 0.f;
    for (int s = 0; s < S; ++s) acc += lg[s] * enc[((size_t)s * B + b) * H + h];
    xbuf[(size_t)b * (E + H) + E + h] = acc;
  }
  for (int e = tid; e < E; e += 128)
    xbuf[(size_t)b * (E + H) + e] = emb_t[(size_t)b * E + e];
}

// ---------------------------------------------------------------------------
// K3 (per step t): GRU cell, one block per batch element, 256 threads
// writes h_new (f32 for recurrence) and f16 copy in WMMA-A fragment order
// ---------------------------------------------------------------------------
__global__ void gru_step_kernel(const float* __restrict__ hprev,
                                const float* __restrict__ xbuf,
                                const float* __restrict__ Wih,  // [3H, E+H]
                                const float* __restrict__ Whh,  // [3H, H]
                                const float* __restrict__ bih,
                                const float* __restrict__ bhh,
                                float* __restrict__ hnew,       // [B,H]
                                _Float16* __restrict__ hf16) {  // [B,KP] packed
  __shared__ float xr[E + H], hr[H];
  const int b = blockIdx.x, tid = threadIdx.x;  // 256 threads
  for (int i = tid; i < E + H; i += 256) xr[i] = xbuf[(size_t)b * (E + H) + i];
  for (int i = tid; i < H; i += 256) hr[i] = hprev[(size_t)b * H + i];
  __syncthreads();
  if (tid < H) {
    const int j = tid;
    float gi[3], gh[3];
#pragma unroll
    for (int g = 0; g < 3; ++g) {
      const float* wi = Wih + (size_t)(g * H + j) * (E + H);
      float acc = bih[g * H + j];
      for (int i = 0; i < E + H; ++i) acc += wi[i] * xr[i];
      gi[g] = acc;
      const float* wh = Whh + (size_t)(g * H + j) * H;
      float acc2 = bhh[g * H + j];
      for (int i = 0; i < H; ++i) acc2 += wh[i] * hr[i];
      gh[g] = acc2;
    }
    const float r = 1.f / (1.f + __expf(-(gi[0] + gh[0])));
    const float z = 1.f / (1.f + __expf(-(gi[1] + gh[1])));
    const float n = tanhf(gi[2] + r * gh[2]);
    const float hn = (1.f - z) * n + z * hr[j];
    hnew[(size_t)b * H + j] = hn;
    hf16[(size_t)b * KP + afrag_off(j)] = (_Float16)hn;
  } else if (tid < KP) {
    hf16[(size_t)b * KP + afrag_off(tid)] = (_Float16)0.f;  // zero K padding
  }
}

// ---------------------------------------------------------------------------
// K4: pack out_W [V,H] -> column(K)-major f16 [NPAD, KP] (zero padded).
// Per-lane B fragment is then 16 contiguous halves => one v16h load.
// ---------------------------------------------------------------------------
__global__ void pack_outW_kernel(const float* __restrict__ outW,
                                 _Float16* __restrict__ Bt) {
  const int idx = blockIdx.x * 256 + threadIdx.x;
  if (idx >= NPAD * KP) return;
  const int n = idx / KP, k = idx % KP;
  const float v = (n < V && k < H) ? outW[(size_t)n * H + k] : 0.f;
  Bt[(size_t)n * KP + afrag_off(k)] = (_Float16)v;
}

// ---------------------------------------------------------------------------
// K5: logits[8192, V] = Hseq @ out_W.T + out_b via v_wmma_f32_16x16x32_f16
// 128 threads = 4 waves; wave w owns a 16x64 strip (rows mTile..+15, 4 n-tiles)
// with 4 accumulators, reusing one A fragment per k-step across 4 WMMAs.
// Block tile = 64m x 64n. Both operands pre-packed: each fragment is one
// contiguous 32B v16h load per lane.
// ---------------------------------------------------------------------------
__global__ void out_gemm_kernel(const _Float16* __restrict__ Af,  // [M, KP] packed
                                const _Float16* __restrict__ Bt,  // [NPAD, KP]
                                const float* __restrict__ bias,   // [V]
                                float* __restrict__ out) {        // [M, V]
  const int tid = threadIdx.x;
  const int w = tid >> 5, lane = tid & 31;
  const int half = lane >> 4, l15 = lane & 15;
  const int mTile = blockIdx.x * 64 + w * 16;
  const int nBase = blockIdx.y * 64;

  const _Float16* arow = Af + (size_t)(mTile + l15) * KP + half * 16;
  const _Float16* bcol = Bt + (size_t)(nBase + l15) * KP + half * 16;
  v8f acc0 = {}, acc1 = {}, acc2 = {}, acc3 = {};
#pragma unroll
  for (int kt = 0; kt < KP; kt += 32) {
    const v16h a = *(const v16h*)(arow + kt);  // fragment-ordered, 32B aligned
    const v16h b0 = *(const v16h*)(bcol + kt);
    const v16h b1 = *(const v16h*)(bcol + 16 * KP + kt);
    const v16h b2 = *(const v16h*)(bcol + 32 * KP + kt);
    const v16h b3 = *(const v16h*)(bcol + 48 * KP + kt);
    acc0 = __builtin_amdgcn_wmma_f32_16x16x32_f16(false, a, false, b0, (short)0, acc0, false, false);
    acc1 = __builtin_amdgcn_wmma_f32_16x16x32_f16(false, a, false, b1, (short)0, acc1, false, false);
    acc2 = __builtin_amdgcn_wmma_f32_16x16x32_f16(false, a, false, b2, (short)0, acc2, false, false);
    acc3 = __builtin_amdgcn_wmma_f32_16x16x32_f16(false, a, false, b3, (short)0, acc3, false, false);
  }
  v8f accs[4] = {acc0, acc1, acc2, acc3};
#pragma unroll
  for (int s2 = 0; s2 < 4; ++s2) {
    const int n = nBase + s2 * 16 + l15;
    if (n < V) {
      const float bv = bias[n];
#pragma unroll
      for (int r = 0; r < 8; ++r) {
        const int m = mTile + half * 8 + r;
        out[(size_t)m * V + n] = accs[s2][r] + bv;
      }
    }
  }
}

// ---------------------------------------------------------------------------
// K6: in-place row log-softmax over V (one block per (t,b) row)
// ---------------------------------------------------------------------------
__global__ void logsoftmax_kernel(float* __restrict__ out) {
  __shared__ float red[256];
  const int row = blockIdx.x, tid = threadIdx.x;
  float* p = out + (size_t)row * V;
  float mx = -3.4e38f;
  for (int n = tid; n < V; n += 256) mx = fmaxf(mx, p[n]);
  red[tid] = mx;
  __syncthreads();
  for (int o = 128; o > 0; o >>= 1) {
    if (tid < o) red[tid] = fmaxf(red[tid], red[tid + o]);
    __syncthreads();
  }
  mx = red[0];
  __syncthreads();
  float sum = 0.f;
  for (int n = tid; n < V; n += 256) sum += __expf(p[n] - mx);
  red[tid] = sum;
  __syncthreads();
  for (int o = 128; o > 0; o >>= 1) {
    if (tid < o) red[tid] += red[tid + o];
    __syncthreads();
  }
  const float lse = mx + __logf(red[0]);
  for (int n = tid; n < V; n += 256) p[n] = p[n] - lse;
}

__global__ void copy_hidden_kernel(const float* __restrict__ src,
                                   float* __restrict__ dst) {
  const int i = blockIdx.x * 256 + threadIdx.x;
  if (i < B * H) dst[i] = src[i];
}

// ---------------------------------------------------------------------------
extern "C" void kernel_launch(void* const* d_in, const int* in_sizes, int n_in,
                              void* d_out, int out_size, void* d_ws, size_t ws_size,
                              hipStream_t stream) {
  const int*   tokens = (const int*)d_in[0];
  const float* hidden = (const float*)d_in[1];
  const float* enc    = (const float*)d_in[2];
  const float* emb    = (const float*)d_in[3];
  const float* W1 = (const float*)d_in[4],  *b1 = (const float*)d_in[5];
  const float* W2 = (const float*)d_in[6],  *b2 = (const float*)d_in[7];
  const float* W3 = (const float*)d_in[8],  *b3 = (const float*)d_in[9];
  const float* attn_W = (const float*)d_in[10], *attn_b = (const float*)d_in[11];
  const float* Wih = (const float*)d_in[12], *Whh = (const float*)d_in[13];
  const float* bih = (const float*)d_in[14], *bhh = (const float*)d_in[15];
  const float* outW = (const float*)d_in[16], *outb = (const float*)d_in[17];
  float* out = (float*)d_out;

  // workspace carve (~31.5 MB)
  char* ws = (char*)d_ws;
  size_t off = 0;
  auto falloc = [&](size_t n) { float* p = (float*)(ws + off); off += n * sizeof(float); return p; };
  float* embedded = falloc((size_t)T * B * E);
  float* ah3      = falloc((size_t)T * B * A);
  float* ah2      = falloc((size_t)B * S * A);
  float* hseq     = falloc((size_t)T * B * H);
  float* xbuf     = falloc((size_t)B * (E + H));
  _Float16* Hf16 = (_Float16*)(ws + off); off += (size_t)T * B * KP * sizeof(_Float16);
  _Float16* Bt   = (_Float16*)(ws + off); off += (size_t)NPAD * KP * sizeof(_Float16);

  const size_t hidOff  = (size_t)T * B * V;
  const size_t attnOff = hidOff + (size_t)B * H;

  // one-time precompute (independent)
  embed_ah3_kernel<<<T * B, 128, 0, stream>>>(tokens, emb, W3, b3, embedded, ah3);
  enc_ah2_kernel<<<dim3(B, S), 128, 0, stream>>>(enc, W2, b2, ah2);
  pack_outW_kernel<<<(NPAD * KP + 255) / 256, 256, 0, stream>>>(outW, Bt);

  // sequential recurrence (cheap part only)
  for (int t = 0; t < T; ++t) {
    const float* hprev = (t == 0) ? hidden : hseq + (size_t)(t - 1) * B * H;
    attn_step_kernel<<<B, 128, 0, stream>>>(
        hprev, embedded + (size_t)t * B * E, ah3 + (size_t)t * B * A,
        ah2, enc, W1, b1, attn_W, attn_b, xbuf,
        out + attnOff + (size_t)t * B * S);
    gru_step_kernel<<<B, 256, 0, stream>>>(
        hprev, xbuf, Wih, Whh, bih, bhh,
        hseq + (size_t)t * B * H, Hf16 + (size_t)t * B * KP);
  }

  // dominant work: one large WMMA GEMM over all timesteps, then log-softmax
  out_gemm_kernel<<<dim3((T * B) / 64, NPAD / 64), 128, 0, stream>>>(Hf16, Bt, outb, out);
  logsoftmax_kernel<<<T * B, 256, 0, stream>>>(out);
  copy_hidden_kernel<<<(B * H + 255) / 256, 256, 0, stream>>>(
      hseq + (size_t)(T - 1) * B * H, out + hidOff);
}
